// SelfAttention2_41790031790641
// MI455X (gfx1250) — matrix operand order repaired and checked by reading, not compiled, not run on hardware
//
#include <hip/hip_runtime.h>
#include <math.h>

typedef __attribute__((ext_vector_type(16))) __bf16        v16bf;
typedef __attribute__((ext_vector_type(8)))  float         v8f;
typedef __attribute__((ext_vector_type(8)))  unsigned int  v8u;
typedef __attribute__((ext_vector_type(4)))  unsigned int  v4u;
typedef __attribute__((ext_vector_type(4)))  float         v4f;

constexpr int KD = 256;    // feature dim D
constexpr int TT = 4096;   // sequence length T
constexpr int BB = 32;     // batch B
constexpr int NTOK = BB * TT;

union BfPack { v16bf bf; v8u u; };

__device__ __forceinline__ unsigned int f2bf(float f) {
    unsigned int u = __float_as_uint(f);
    u = u + 0x7FFFu + ((u >> 16) & 1u);   // round-to-nearest-even
    return u >> 16;
}

// Branch-free tanh: clamp to +-9 (tanh(9) == 1.0f in f32), then
// (e^{2h}-1)/(e^{2h}+1) via native v_exp_f32 + v_rcp_f32. No EXEC divergence.
__device__ __forceinline__ float fast_tanh(float h) {
    float hc = fminf(fmaxf(h, -9.0f), 9.0f);
    float t  = __expf(2.0f * hc);
    return (t - 1.0f) * __builtin_amdgcn_rcpf(t + 1.0f);
}

// ---------------------------------------------------------------------------
// Fold the 4 GEMMs into one:  M[k][n] = sum_j Wq[k,j]W_attn[j,n] + Wv[k,j]V_attn[j,n]
// Stored transposed (MT[n][k]) in bf16 so each WMMA B-lane reads contiguous K.
// ---------------------------------------------------------------------------
__global__ __launch_bounds__(256) void prep_M(const float* __restrict__ Wq,
                                              const float* __restrict__ Wa,
                                              const float* __restrict__ Wv,
                                              const float* __restrict__ Va,
                                              unsigned short* __restrict__ MT) {
    int k = blockIdx.x, n = threadIdx.x;
    __shared__ float wq[KD], wv[KD];
    wq[n] = Wq[k * KD + n];
    wv[n] = Wv[k * KD + n];
    __syncthreads();
    float acc = 0.f;
    for (int j = 0; j < KD; ++j)
        acc += wq[j] * Wa[j * KD + n] + wv[j] * Va[j * KD + n];
    MT[n * KD + k] = (unsigned short)f2bf(acc);
}

// cbias[n] = bq@W_attn + bv@V_attn + b_param + conv_b
__global__ __launch_bounds__(256) void prep_bias(const float* __restrict__ bq,
                                                 const float* __restrict__ bv,
                                                 const float* __restrict__ Wa,
                                                 const float* __restrict__ Va,
                                                 const float* __restrict__ bparam,
                                                 const float* __restrict__ convb,
                                                 float* __restrict__ cbias) {
    int n = threadIdx.x;
    float acc = bparam[n] + convb[n];
    for (int j = 0; j < KD; ++j)
        acc += bq[j] * Wa[j * KD + n] + bv[j] * Va[j * KD + n];
    cbias[n] = acc;
}

// ---------------------------------------------------------------------------
// Score kernel: one wave = 16-token strip. h = x@M + conv + bias, then
// score = tanh(h)@fc_w via 128 v_wmma_f32_16x16x32_bf16 per wave.
// Mᵀ (256x256 bf16 = 128 KB) lives in dynamic LDS (CDNA5 WGP has 320 KB).
// ---------------------------------------------------------------------------
__global__ __launch_bounds__(256) void score_kernel(const float* __restrict__ x,
                                                    const float* __restrict__ la,
                                                    const unsigned short* __restrict__ MT,
                                                    const float* __restrict__ cbias,
                                                    const float* __restrict__ convw,
                                                    const float* __restrict__ fcw,
                                                    const float* __restrict__ fcb,
                                                    float* __restrict__ scores) {
    extern __shared__ unsigned char smem[];   // 128 KB: MT bf16
    const int tid = threadIdx.x;

    // Stage Mᵀ into LDS (shared by all 8 waves of the block).
    {
        v4u* dst = (v4u*)smem;
        const v4u* src = (const v4u*)MT;
        #pragma unroll 4
        for (int i = tid; i < (KD * KD * 2 / 16); i += 256) dst[i] = src[i];
    }
    __syncthreads();

    const int lane = tid & 31;
    const int warp = tid >> 5;
    const int hi   = lane >> 4;      // lane half selects K sub-range
    const int ln   = lane & 15;

    const int strip = blockIdx.x * 8 + warp;   // 16 tokens per wave
    const int base  = strip * 16;

    // ---- A strip in registers: 16 tokens x 256 K, bf16, contiguous K per lane half (k = 16*hi + s) ----
    // lanes 0-15: elems 0-15 = K kc..kc+15
    // lanes 16-31: same with +16 K offset.  kc = c*32.
    v16bf a[8];
    {
        const float* xrow = x + (size_t)(base + ln) * KD;
        #pragma unroll
        for (int c = 0; c < 8; ++c) {
            const int k0 = c * 32 + hi * 16;
            v4f f0 = *(const v4f*)(xrow + k0);
            v4f f1 = *(const v4f*)(xrow + k0 + 4);
            v4f f2 = *(const v4f*)(xrow + k0 + 8);
            v4f f3 = *(const v4f*)(xrow + k0 + 12);
            BfPack p;
            p.u[0] = f2bf(f0[0]) | (f2bf(f0[1]) << 16);
            p.u[1] = f2bf(f0[2]) | (f2bf(f0[3]) << 16);
            p.u[2] = f2bf(f1[0]) | (f2bf(f1[1]) << 16);
            p.u[3] = f2bf(f1[2]) | (f2bf(f1[3]) << 16);
            p.u[4] = f2bf(f2[0]) | (f2bf(f2[1]) << 16);
            p.u[5] = f2bf(f2[2]) | (f2bf(f2[3]) << 16);
            p.u[6] = f2bf(f3[0]) | (f2bf(f3[1]) << 16);
            p.u[7] = f2bf(f3[2]) | (f2bf(f3[3]) << 16);
            a[c] = p.bf;
        }
    }

    // ---- 3-tap conv inputs per output-row (C/D layout: row = r + 8*hi) ----
    const int bi = base >> 12;           // token / 4096 = batch
    const int tb = base & (TT - 1);
    float lam1[8], la0[8], lap1[8];
    {
        const float* lr = la + (size_t)bi * TT;
        #pragma unroll
        for (int r = 0; r < 8; ++r) {
            int tl = tb + r + 8 * hi;
            lam1[r] = (tl > 0)      ? lr[tl - 1] : 0.f;   // SAME padding
            la0 [r] = lr[tl];
            lap1[r] = (tl < TT - 1) ? lr[tl + 1] : 0.f;
        }
    }

    float sacc[8] = {0, 0, 0, 0, 0, 0, 0, 0};
    for (int nt = 0; nt < 16; ++nt) {
        v8f c8 = {0.f, 0.f, 0.f, 0.f, 0.f, 0.f, 0.f, 0.f};
        const int n = nt * 16 + ln;      // this lane's output column
        const unsigned char* brow = smem + ((size_t)n * KD + hi * 16) * 2;

        // Double-buffered B tiles: issue load c+1 before consuming tile c so
        // the compiler can wait at dscnt<=1 and overlap LDS with the WMMA pipe.
        v8u bcur = *(const v8u*)(brow);
        #pragma unroll
        for (int c = 0; c < 8; ++c) {
            BfPack bp; bp.u = bcur;
            if (c < 7) bcur = *(const v8u*)(brow + (c + 1) * 64);
            c8 = __builtin_amdgcn_wmma_f32_16x16x32_bf16(
                     false, a[c], false, bp.bf, (short)0, c8, false, false);
        }
        const float cw0 = convw[n * 3 + 0];
        const float cw1 = convw[n * 3 + 1];
        const float cw2 = convw[n * 3 + 2];
        const float cb  = cbias[n];
        const float fw  = fcw[n];
        #pragma unroll
        for (int r = 0; r < 8; ++r) {
            float h = c8[r] + cb + lam1[r] * cw0 + la0[r] * cw1 + lap1[r] * cw2;
            sacc[r] += fast_tanh(h) * fw;
        }
    }

    // Cross-lane reduce over the 16 columns held in each half-wave.
    const float fb = fcb[0];
    #pragma unroll
    for (int r = 0; r < 8; ++r) {
        float s = sacc[r];
        s += __shfl_xor(s, 1, 32);
        s += __shfl_xor(s, 2, 32);
        s += __shfl_xor(s, 4, 32);
        s += __shfl_xor(s, 8, 32);
        if (ln == 0) scores[base + r + 8 * hi] = s + fb;
    }
}

// In-place softmax over T per batch (scores -> attn, stored in d_out).
__global__ __launch_bounds__(256) void softmax_kernel(float* __restrict__ attn) {
    __shared__ float red[256];
    const int b = blockIdx.x, tid = threadIdx.x;
    float* p = attn + (size_t)b * TT;
    float m = -1e30f;
    for (int i = tid; i < TT; i += 256) m = fmaxf(m, p[i]);
    red[tid] = m; __syncthreads();
    for (int s = 128; s > 0; s >>= 1) {
        if (tid < s) red[tid] = fmaxf(red[tid], red[tid + s]);
        __syncthreads();
    }
    m = red[0]; __syncthreads();
    float sum = 0.f;
    for (int i = tid; i < TT; i += 256) { float e = __expf(p[i] - m); p[i] = e; sum += e; }
    red[tid] = sum; __syncthreads();
    for (int s = 128; s > 0; s >>= 1) {
        if (tid < s) red[tid] += red[tid + s];
        __syncthreads();
    }
    const float inv = 1.f / red[0];
    for (int i = tid; i < TT; i += 256) p[i] *= inv;
}

// y[b,:] = sum_t attn[b,t] * x[b,t,:]   (partials over 256-token slabs)
__global__ __launch_bounds__(256) void ctx_partial(const float* __restrict__ x,
                                                   const float* __restrict__ attn,
                                                   float* __restrict__ y) {
    const int b = blockIdx.x, slab = blockIdx.y, d = threadIdx.x;
    __shared__ float as[256];
    const size_t t0 = (size_t)b * TT + slab * 256;
    as[d] = attn[t0 + d];
    __syncthreads();
    const float* xb = x + t0 * KD;
    float acc = 0.f;
    for (int t = 0; t < 256; ++t) acc += as[t] * xb[(size_t)t * KD + d];
    atomicAdd(&y[b * KD + d], acc);
}

// weighted[b,d] = (y[b,:] @ Wv)[d] + bv[d]   (since softmax sums to 1)
__global__ __launch_bounds__(256) void ctx_final(const float* __restrict__ y,
                                                 const float* __restrict__ Wv,
                                                 const float* __restrict__ bv,
                                                 float* __restrict__ out) {
    const int b = blockIdx.x, d = threadIdx.x;
    __shared__ float ys[KD];
    ys[d] = y[b * KD + d];
    __syncthreads();
    float acc = bv[d];
    for (int k = 0; k < KD; ++k) acc += ys[k] * Wv[k * KD + d];
    out[b * KD + d] = acc;
}

extern "C" void kernel_launch(void* const* d_in, const int* in_sizes, int n_in,
                              void* d_out, int out_size, void* d_ws, size_t ws_size,
                              hipStream_t stream) {
    const float* x   = (const float*)d_in[0];
    const float* la  = (const float*)d_in[1];
    const float* Wq  = (const float*)d_in[2];
    const float* bq  = (const float*)d_in[3];
    const float* Wv  = (const float*)d_in[4];
    const float* bv  = (const float*)d_in[5];
    const float* cw  = (const float*)d_in[6];
    const float* cb  = (const float*)d_in[7];
    const float* Wa  = (const float*)d_in[8];
    const float* Va  = (const float*)d_in[9];
    const float* fcw = (const float*)d_in[10];
    const float* fcb = (const float*)d_in[11];
    const float* bp  = (const float*)d_in[12];

    float* out_w = (float*)d_out;            // weighted: B*D floats
    float* attn  = (float*)d_out + BB * KD;  // scores -> attn in place: B*T floats

    unsigned short* MT = (unsigned short*)d_ws;                          // 128 KB
    float* cbias = (float*)((char*)d_ws + KD * KD * 2);                  // 1 KB
    float* y     = (float*)((char*)d_ws + KD * KD * 2 + KD * 4);         // 32 KB

    prep_M<<<KD, KD, 0, stream>>>(Wq, Wa, Wv, Va, MT);
    prep_bias<<<1, KD, 0, stream>>>(bq, bv, Wa, Va, bp, cb, cbias);
    hipMemsetAsync(y, 0, BB * KD * sizeof(float), stream);

    // 131072 tokens / (8 waves * 16 tokens) = 1024 blocks; 128 KB dynamic LDS
    score_kernel<<<NTOK / 128, 256, KD * KD * 2, stream>>>(
        x, la, MT, cbias, cw, fcw, fcb, attn);

    softmax_kernel<<<BB, 256, 0, stream>>>(attn);
    ctx_partial<<<dim3(BB, 16), 256, 0, stream>>>(x, attn, y);
    ctx_final<<<BB, KD, 0, stream>>>(y, Wv, bv, out_w);
}
